// DecoderGRUCell_7610682048614
// MI455X (gfx1250) — matrix operand order repaired and checked by reading, not compiled, not run on hardware
//
#include <hip/hip_runtime.h>
#include <cstdint>
#include <cstddef>

// Problem dims (fixed by reference)
#define BB 64
#define TT 256
#define EE 1024
#define UU 1024

typedef __attribute__((ext_vector_type(16))) __bf16 v16bf;
typedef __attribute__((ext_vector_type(8)))  float  v8f;

// ---------- helpers ----------
__device__ __forceinline__ __bf16 f2bf(float f) {
  union { float f; uint32_t u; } v; v.f = f;
  uint32_t u = v.u;
  u += 0x7fffu + ((u >> 16) & 1u);          // round-to-nearest-even
  uint16_t h = (uint16_t)(u >> 16);
  __bf16 r; __builtin_memcpy(&r, &h, 2);
  return r;
}

__device__ __forceinline__ float hsig(float x) {
  return fminf(fmaxf(0.2f * x + 0.5f, 0.0f), 1.0f);
}

// Packed WMMA-B layout index: B[k][n] (k up K, n across N, N multiple of 16)
// -> [kt = k/32][nt = n/16][lane][i]  where lane = ((k%32)>=16 ? 16:0) + n%16,
//    i = k%16. Per-lane 16 bf16 are contiguous (32 B) => 2x b128 coalesced load.
__device__ __forceinline__ size_t bpk_idx(int k, int n, int N) {
  int kt = k >> 5, kr = k & 31;
  int nt = n >> 4, nr = n & 15;
  int lane = ((kr >> 4) ? 16 : 0) + nr;
  int i = kr & 15;
  return (((size_t)kt * (N >> 4) + nt) * 32 + lane) * 16 + i;
}

// ---------- WMMA A-fragment loader (contiguous-K layout, wave32) ----------
#define LDA 40    // 32 + 8 pad (bf16 elems)

// A fragment: 16x32 bf16 tile, row-major in LDS with stride LDA.
// lane L: M = L%16 ; K-groups: lanes 0-15 -> K {0..15}, lanes 16-31 -> K {16..31}
// (16 contiguous elements -> 2x ds_load_b128)
__device__ __forceinline__ v16bf load_a_frag(const __bf16* As) {
  const int lane = threadIdx.x & 31;
  const int m  = lane & 15;
  const int kb = (lane < 16) ? 0 : 16;
  v16bf a;
#pragma unroll
  for (int i = 0; i < 16; ++i) {
    int k = kb + i;
    a[i] = As[m * LDA + k];
  }
  return a;
}

// ---------- block GEMM core: C(64 x 128) += A(64 x Ka) * Bpacked ----------
// 256 threads = 8 waves (4 M-waves x 2 N-waves); wave tile 16x64 (4 x v8f accum).
// A staged in LDS 128-K-deep per barrier pair (4 sub-chunks => 16 WMMA / barrier).
// Bpk is in packed fragment layout (see bpk_idx); NtB = N/16 of the B matrix.
__device__ __forceinline__ void gemm_block_64x128(
    const __bf16* __restrict__ Aglob, int Ka, int m0,
    const v16bf* __restrict__ Bpk, int NtB, int n0,
    v8f acc[4])
{
  __shared__ __bf16 As[4 * 64 * LDA];   // 4 sub-chunks of 64x32, 20 KB

  const int tid  = threadIdx.x;
  const int wave = tid >> 5;
  const int wm   = wave & 3;     // 0..3 : M sub-tile
  const int wn   = wave >> 2;    // 0..1 : N sub-tile
  const int lane = tid & 31;
  const int ntbase = (n0 >> 4) + wn * 4;   // first 16-col tile of this wave

#pragma unroll
  for (int j = 0; j < 4; ++j)
#pragma unroll
    for (int v = 0; v < 8; ++v) acc[j][v] = 0.0f;

  for (int k0 = 0; k0 < Ka; k0 += 128) {
    // stage A 64x128: each thread copies 64 contiguous bytes of one row
    {
      int m = tid >> 2;          // 0..63
      int c = tid & 3;           // sub-chunk 0..3
      const __bf16* src = &Aglob[(size_t)(m0 + m) * Ka + k0 + c * 32];
      __bf16* dst = &As[(c * 64 + m) * LDA];
#pragma unroll
      for (int q = 0; q < 4; ++q)
        *(uint4*)&dst[q * 8] = *(const uint4*)&src[q * 8];
    }
    __syncthreads();

#pragma unroll
    for (int c = 0; c < 4; ++c) {
      v16bf af = load_a_frag(&As[(c * 64 + wm * 16) * LDA]);
      const size_t krow = (size_t)((k0 >> 5) + c) * NtB;
      v16bf b0 = Bpk[(krow + ntbase + 0) * 32 + lane];
      v16bf b1 = Bpk[(krow + ntbase + 1) * 32 + lane];
      v16bf b2 = Bpk[(krow + ntbase + 2) * 32 + lane];
      v16bf b3 = Bpk[(krow + ntbase + 3) * 32 + lane];
      acc[0] = __builtin_amdgcn_wmma_f32_16x16x32_bf16(
          false, af, false, b0, (short)0, acc[0], false, false);
      acc[1] = __builtin_amdgcn_wmma_f32_16x16x32_bf16(
          false, af, false, b1, (short)0, acc[1], false, false);
      acc[2] = __builtin_amdgcn_wmma_f32_16x16x32_bf16(
          false, af, false, b2, (short)0, acc[2], false, false);
      acc[3] = __builtin_amdgcn_wmma_f32_16x16x32_bf16(
          false, af, false, b3, (short)0, acc[3], false, false);
    }
    __syncthreads();
  }
}

// ---------- prep kernels ----------
__global__ void k_cast_bf16(const float* __restrict__ src, __bf16* __restrict__ dst, size_t n) {
  size_t i = (size_t)blockIdx.x * blockDim.x + threadIdx.x;
  size_t st = (size_t)gridDim.x * blockDim.x;
  for (; i < n; i += st) dst[i] = f2bf(src[i]);
}

// Wcat (packed): logical (K=2048) x (N=3072); rows 0..1023 = C_g, 1024..2047 = W_g;
// gates [r|z|cand] along N
__global__ void k_build_wcat(const float* __restrict__ Wr, const float* __restrict__ Cr,
                             const float* __restrict__ Wz, const float* __restrict__ Cz,
                             const float* __restrict__ W,  const float* __restrict__ C,
                             __bf16* __restrict__ Wpk) {
  size_t i = (size_t)blockIdx.x * blockDim.x + threadIdx.x;
  size_t st = (size_t)gridDim.x * blockDim.x;
  const size_t n = (size_t)2048 * 3072;
  for (; i < n; i += st) {
    int k = (int)(i / 3072), c = (int)(i % 3072);
    int g = c >> 10, nn = c & 1023;
    const float* top = (g == 0) ? Cr : (g == 1) ? Cz : C;   // ctx block (K 0..1023)
    const float* bot = (g == 0) ? Wr : (g == 1) ? Wz : W;   // emb block (K 1024..2047)
    float v = (k < 1024) ? top[(size_t)k * 1024 + nn]
                         : bot[(size_t)(k - 1024) * 1024 + nn];
    Wpk[bpk_idx(k, c, 3072)] = f2bf(v);
  }
}

// Ucat2 (packed): logical (1024 x 2048) = [U_r | U_z]; Ubf (packed): (1024 x 1024) = U
__global__ void k_build_u(const float* __restrict__ Ur, const float* __restrict__ Uz,
                          const float* __restrict__ U,
                          __bf16* __restrict__ Upk2, __bf16* __restrict__ Upk) {
  size_t i = (size_t)blockIdx.x * blockDim.x + threadIdx.x;
  size_t st = (size_t)gridDim.x * blockDim.x;
  const size_t n = (size_t)1024 * 3072;
  for (; i < n; i += st) {
    int k = (int)(i / 3072), c = (int)(i % 3072);
    if (c < 1024)
      Upk2[bpk_idx(k, c, 2048)] = f2bf(Ur[(size_t)k * 1024 + c]);
    else if (c < 2048)
      Upk2[bpk_idx(k, c, 2048)] = f2bf(Uz[(size_t)k * 1024 + (c - 1024)]);
    else
      Upk[bpk_idx(k, c - 2048, 1024)] = f2bf(U[(size_t)k * 1024 + (c - 2048)]);
  }
}

__global__ void k_init_h(const float* __restrict__ h0, float* __restrict__ h,
                         __bf16* __restrict__ hbf) {
  size_t i = (size_t)blockIdx.x * blockDim.x + threadIdx.x;
  if (i < (size_t)BB * UU) { h[i] = h0[i]; hbf[i] = f2bf(h0[i]); }
}

// ---------- big precompute GEMM: G(16384 x 3072) = Xbf(16384 x 2048) @ Wcat ----------
__global__ void __launch_bounds__(256)
k_gemm_pre(const __bf16* __restrict__ X, const v16bf* __restrict__ Wpk,
           float* __restrict__ G) {
  const int n0 = blockIdx.x * 128;
  const int m0 = blockIdx.y * 64;
  v8f acc[4];
  gemm_block_64x128(X, 2048, m0, Wpk, 3072 >> 4, n0, acc);

  const int tid = threadIdx.x, wave = tid >> 5, wm = wave & 3, wn = wave >> 2;
  const int lane = tid & 31, cn = lane & 15, cm = (lane < 16) ? 0 : 8;
#pragma unroll
  for (int j = 0; j < 4; ++j)
#pragma unroll
    for (int v = 0; v < 8; ++v) {
      int m = m0 + wm * 16 + cm + v;
      int n = n0 + wn * 64 + j * 16 + cn;
      G[(size_t)m * 3072 + n] = acc[j][v];
    }
}

// ---------- recurrence phase A: S = h @ [U_r|U_z] ; r,z gates ; hr = h*r ----------
__global__ void __launch_bounds__(256)
k_phase_a(const __bf16* __restrict__ hbf, const v16bf* __restrict__ Upk2,
          const float* __restrict__ G, const float* __restrict__ h,
          __bf16* __restrict__ hrbf, float* __restrict__ zbuf, int t) {
  const int n0 = blockIdx.x * 128;   // 16 blocks -> N = 2048
  v8f acc[4];
  gemm_block_64x128(hbf, 1024, 0, Upk2, 2048 >> 4, n0, acc);

  const int tid = threadIdx.x, wave = tid >> 5, wm = wave & 3, wn = wave >> 2;
  const int lane = tid & 31, cn = lane & 15, cm = (lane < 16) ? 0 : 8;
#pragma unroll
  for (int j = 0; j < 4; ++j)
#pragma unroll
    for (int v = 0; v < 8; ++v) {
      int b  = wm * 16 + cm + v;                 // batch row (0..63)
      int gn = n0 + wn * 64 + j * 16 + cn;       // 0..2047
      size_t grow = ((size_t)b * TT + t) * 3072;
      if (gn < 1024) {                           // gate r -> write hr = h * r
        float r  = hsig(acc[j][v] + G[grow + gn]);
        float hv = h[(size_t)b * 1024 + gn];
        hrbf[(size_t)b * 1024 + gn] = f2bf(hv * r);
      } else {                                   // gate z -> stash fp32
        int c = gn - 1024;
        zbuf[(size_t)b * 1024 + c] = hsig(acc[j][v] + G[grow + 1024 + c]);
      }
    }
}

// ---------- recurrence phase B: hc = tanh(Gc + hr @ U) ; h = (1-z)h + z*hc ----------
__global__ void __launch_bounds__(256)
k_phase_b(const __bf16* __restrict__ hrbf, const v16bf* __restrict__ Upk,
          const float* __restrict__ G, float* __restrict__ h,
          __bf16* __restrict__ hbf, const float* __restrict__ zbuf,
          float* __restrict__ out, int t) {
  const int n0 = blockIdx.x * 128;   // 8 blocks -> N = 1024
  v8f acc[4];
  gemm_block_64x128(hrbf, 1024, 0, Upk, 1024 >> 4, n0, acc);

  const int tid = threadIdx.x, wave = tid >> 5, wm = wave & 3, wn = wave >> 2;
  const int lane = tid & 31, cn = lane & 15, cm = (lane < 16) ? 0 : 8;
#pragma unroll
  for (int j = 0; j < 4; ++j)
#pragma unroll
    for (int v = 0; v < 8; ++v) {
      int b = wm * 16 + cm + v;
      int n = n0 + wn * 64 + j * 16 + cn;
      size_t bi = (size_t)b * 1024 + n;
      float hc = tanhf(acc[j][v] + G[((size_t)b * TT + t) * 3072 + 2048 + n]);
      float z  = zbuf[bi];
      float hn = (1.0f - z) * h[bi] + z * hc;
      h[bi]   = hn;
      hbf[bi] = f2bf(hn);
      out[((size_t)b * TT + t) * 1024 + n] = hn;
    }
}

// ---------- host launcher ----------
extern "C" void kernel_launch(void* const* d_in, const int* in_sizes, int n_in,
                              void* d_out, int out_size, void* d_ws, size_t ws_size,
                              hipStream_t stream) {
  (void)in_sizes; (void)n_in; (void)out_size; (void)ws_size;
  const float* all_in = (const float*)d_in[0];
  const float* h0     = (const float*)d_in[1];
  const float* Wr = (const float*)d_in[2];
  const float* Ur = (const float*)d_in[3];
  const float* Cr = (const float*)d_in[4];
  const float* Wz = (const float*)d_in[5];
  const float* Uz = (const float*)d_in[6];
  const float* Cz = (const float*)d_in[7];
  const float* W  = (const float*)d_in[8];
  const float* U  = (const float*)d_in[9];
  const float* C  = (const float*)d_in[10];
  float* out = (float*)d_out;

  // workspace layout (bytes); total ~288 MB
  char* ws = (char*)d_ws;
  __bf16* Xbf  = (__bf16*)(ws + 0);                    //  67,108,864 B
  __bf16* Wpk  = (__bf16*)(ws + 67108864);             //  12,582,912 B (packed)
  __bf16* Upk2 = (__bf16*)(ws + 79691776);             //   4,194,304 B (packed)
  __bf16* Upk  = (__bf16*)(ws + 83886080);             //   2,097,152 B (packed)
  float*  G    = (float*) (ws + 85983232);             // 201,326,592 B
  float*  h    = (float*) (ws + 287309824);            //     262,144 B
  __bf16* hbf  = (__bf16*)(ws + 287571968);            //     131,072 B
  __bf16* hrbf = (__bf16*)(ws + 287703040);            //     131,072 B
  float*  zbuf = (float*) (ws + 287834112);            //     262,144 B

  const size_t nX = (size_t)BB * TT * (EE + UU);       // 33,554,432

  // prep: bf16 conversions + fragment-layout weight packing
  k_cast_bf16<<<8192, 256, 0, stream>>>(all_in, Xbf, nX);
  k_build_wcat<<<4096, 256, 0, stream>>>(Wr, Cr, Wz, Cz, W, C, Wpk);
  k_build_u<<<4096, 256, 0, stream>>>(Ur, Uz, U, Upk2, Upk);
  k_init_h<<<256, 256, 0, stream>>>(h0, h, hbf);

  // hoisted input GEMM: G = X @ Wcat   (16384 x 2048 x 3072)
  k_gemm_pre<<<dim3(3072 / 128, 16384 / 64), 256, 0, stream>>>(
      Xbf, (const v16bf*)Wpk, G);

  // sequential recurrence
  for (int t = 0; t < TT; ++t) {
    k_phase_a<<<2048 / 128, 256, 0, stream>>>(
        hbf, (const v16bf*)Upk2, G, h, hrbf, zbuf, t);
    k_phase_b<<<1024 / 128, 256, 0, stream>>>(
        (const __bf16*)hrbf, (const v16bf*)Upk, G, h, hbf, zbuf, out, t);
  }
}
